// Decoder_48627619725347
// MI455X (gfx1250) — matrix operand order repaired and checked by reading, not compiled, not run on hardware
//
#include <hip/hip_runtime.h>

// ---------------------------------------------------------------------------
// LSTM caption decoder on gfx1250 (MI455X), folded-recurrence persistent
// version:
//   x_{t+1} = h_{t+1} @ W_ho^T + b_ho feeds straight into the next input
//   projection, so the recurrence collapses to ONE K=512 GEMM per step:
//       gates_t = h_t @ W_eff^T + b_eff,   W_eff = W_ih @ W_ho + W_hh
//   (step 0 uses W_hh with bias0 = b + W_ih @ sos).  Logits for all T steps
//   are computed AFTER the recurrence as one 51200x512x512 WMMA GEMM.
//   - persistent kernel: 1 grid barrier per step (atomic + acquire spin)
//   - per-block weight slices (~130 KB bf16) staged once into 320KB/WGP LDS;
//     W_hh slice staged via global_load_async_to_lds_b128 (ASYNCcnt path)
//   - all GEMMs: v_wmma_f32_16x16x32_bf16, fp32 accumulate
// ---------------------------------------------------------------------------

typedef __attribute__((ext_vector_type(16))) __bf16 v16bf;
typedef __attribute__((ext_vector_type(8)))  __bf16 v8bf;
typedef __attribute__((ext_vector_type(8)))  float  v8f;

#define BSZ  512
#define ENCD 2048
#define ED   512
#define HD   512
#define GW   2048        // 4*H gate width
#define GRIDN 128        // persistent grid (32 n-slices x 4 m-groups)

#define WSL_LD 520       // LDS leading dim for 512-wide weight rows (+8 pad)
#define SMEM_ELEMS (2 * 64 * WSL_LD)   // W_eff slice + W_hh slice

// ---- helpers ---------------------------------------------------------------

__device__ __forceinline__ unsigned short f2bf(float f) {
    unsigned u = __float_as_uint(f);
    u += 0x7FFFu + ((u >> 16) & 1u);          // round-to-nearest-even
    return (unsigned short)(u >> 16);
}

__device__ __forceinline__ float sigm(float x) {
    return 1.0f / (1.0f + __expf(-x));
}

__device__ __forceinline__ float tanh_fast(float x) {
    float e = __expf(-2.0f * fabsf(x));
    float r = (1.0f - e) / (1.0f + e);
    return copysignf(r, x);
}

// A fragment: 16x32 bf16 tile (row-major, leading dim lda), per ISA 16-bit
// A layout: lane L<16 -> row m_base+L, K {0..7}U{16..23}; lane>=16 -> K +8.
__device__ __forceinline__ v16bf load_frag_a(const unsigned short* A, int lda,
                                             int m_base, int k0, int lane) {
    int row  = m_base + (lane & 15);
    int koff = (lane >> 4) ? 8 : 0;
    const unsigned short* p = A + (size_t)row * lda + k0 + koff;
    v8bf lo = *(const v8bf*)(p);
    v8bf hi = *(const v8bf*)(p + 16);
    return __builtin_shufflevector(lo, hi, 0,1,2,3,4,5,6,7,8,9,10,11,12,13,14,15);
}

// B fragment: 16 rows of W (row-major, leading dim ldb), contiguous K.
// Lane L owns column n = row_base+(L&15), K {0..7}U{16..23} (L>=16 ? +8 : 0).
// Works for global or LDS pointers (addrspace inferred after inlining).
__device__ __forceinline__ v16bf load_frag_b(const unsigned short* W, int ldb,
                                             int row_base, int k0, int lane) {
    int row = row_base + (lane & 15);
    int kb  = k0 + ((lane >> 4) << 3);
    const unsigned short* p = W + (size_t)row * ldb + kb;
    v8bf lo = *(const v8bf*)(p);
    v8bf hi = *(const v8bf*)(p + 16);
    return __builtin_shufflevector(lo, hi, 0,1,2,3,4,5,6,7,8,9,10,11,12,13,14,15);
}

__device__ __forceinline__ v8f wmma_bf16(v8f acc, v16bf a, v16bf b) {
    return __builtin_amdgcn_wmma_f32_16x16x32_bf16(false, a, false, b,
                                                   (short)0, acc, false, false);
}

// Device-wide sense barrier (monotonic counter, release/acquire).
__device__ __forceinline__ void grid_barrier(unsigned* bar, unsigned target) {
    __threadfence();                 // release: each thread flushes its stores
    __syncthreads();
    if (threadIdx.x == 0) {
        atomicAdd(bar, 1u);
        while (__hip_atomic_load(bar, __ATOMIC_ACQUIRE,
                                 __HIP_MEMORY_SCOPE_AGENT) < target) {
            __builtin_amdgcn_s_sleep(1);
        }
    }
    __syncthreads();                 // acquire-invalidate covers whole WGP
}

// ---- precompute kernels ----------------------------------------------------

__global__ void k_f2bf(const float* __restrict__ in, unsigned short* __restrict__ out, int n) {
    int i = blockIdx.x * 256 + threadIdx.x;
    if (i < n) out[i] = f2bf(in[i]);
}

// transpose-convert a (512,512) f32 matrix to bf16 (out = in^T)
__global__ void k_f2bf_t(const float* __restrict__ in, unsigned short* __restrict__ out) {
    int i = blockIdx.x * 256 + threadIdx.x;
    if (i < HD * ED) {
        int r = i >> 9, cidx = i & 511;
        out[(size_t)cidx * HD + r] = f2bf(in[i]);
    }
}

// b_eff = b_ih + b_hh + W_ih @ b_ho ;  bias0 = b_ih + b_hh + W_ih @ sos
__global__ void k_bias2(const float* __restrict__ Wih, const float* __restrict__ bih,
                        const float* __restrict__ bhh, const float* __restrict__ bho,
                        const float* __restrict__ sos,
                        float* __restrict__ b_eff, float* __restrict__ bias0) {
    int i = blockIdx.x * 256 + threadIdx.x;
    if (i < GW) {
        float s1 = 0.f, s2 = 0.f;
        for (int j = 0; j < ED; ++j) {
            float w = Wih[(size_t)i * ED + j];
            s1 += w * bho[j];
            s2 += w * sos[j];
        }
        float bb = bih[i] + bhh[i];
        b_eff[i] = bb + s1;
        bias0[i] = bb + s2;
    }
}

__global__ void k_init_bar(unsigned* bar) { if (threadIdx.x == 0) bar[0] = 0u; }

// W_eff = W_ih @ W_ho + W_hh   (2048 x 512, bf16 out, f32 accumulate)
__global__ void k_weff(const unsigned short* __restrict__ Wihb,   // (2048,512) bf16
                       const unsigned short* __restrict__ Whot,   // (512,512) bf16 = W_ho^T
                       const float* __restrict__ Whh,             // (2048,512) f32
                       unsigned short* __restrict__ Weff) {       // (2048,512) bf16
    int lane = threadIdx.x & 31, wave = threadIdx.x >> 5;
    int n_base = blockIdx.x * 16;
    int m_base = (blockIdx.y * 8 + wave) * 16;
    v8f acc = {};
    for (int k0 = 0; k0 < ED; k0 += 32) {
        v16bf a = load_frag_a(Wihb, ED, m_base, k0, lane);
        v16bf b = load_frag_b(Whot, ED, n_base, k0, lane);
        acc = wmma_bf16(acc, a, b);
    }
    int n = n_base + (lane & 15);
    int mrow = m_base + ((lane >> 4) << 3);
#pragma unroll
    for (int r = 0; r < 8; ++r) {
        int m = mrow + r;
        Weff[(size_t)m * HD + n] = f2bf(acc[r] + Whh[(size_t)m * HD + n]);
    }
}

// h0 = img @ W_out^T + b_out ; writes c=h0 (f32) and Hall[0] (bf16)
__global__ void k_h0(const unsigned short* __restrict__ imgb,
                     const unsigned short* __restrict__ Woutb,
                     const float* __restrict__ b_out,
                     float* __restrict__ c,
                     unsigned short* __restrict__ Hall0) {
    int lane = threadIdx.x & 31, wave = threadIdx.x >> 5;
    int n_base = blockIdx.x * 16;
    int m_base = (blockIdx.y * 8 + wave) * 16;
    v8f acc = {};
    for (int k0 = 0; k0 < ENCD; k0 += 32) {
        v16bf a = load_frag_a(imgb, ENCD, m_base, k0, lane);
        v16bf b = load_frag_b(Woutb, ENCD, n_base, k0, lane);
        acc = wmma_bf16(acc, a, b);
    }
    int n = n_base + (lane & 15);
    float bias = b_out[n];
    int mrow = m_base + ((lane >> 4) << 3);
#pragma unroll
    for (int r = 0; r < 8; ++r) {
        float v = acc[r] + bias;
        int m = mrow + r;
        c[(size_t)m * HD + n] = v;
        Hall0[(size_t)m * HD + n] = f2bf(v);
    }
}

// ---- persistent recurrence kernel ------------------------------------------
// Block b: n-slice nb = b&31, m-group mg = b>>5; wave w owns the 16x16 cell
// tile (m_tile = mg*8+w, nb) for all T steps.  One GEMM + one barrier / step.
__global__ void k_persist(const unsigned short* __restrict__ Weff,  // (2048,512) bf16
                          const unsigned short* __restrict__ Whh,   // (2048,512) bf16
                          const float* __restrict__ b_eff,          // (2048) f32
                          const float* __restrict__ bias0,          // (2048) f32
                          float* __restrict__ c,                    // (512,512) f32
                          unsigned short* __restrict__ Hall,        // (T+1,512,512) bf16
                          int T, unsigned* __restrict__ bar) {
    extern __shared__ unsigned short smem[];
    unsigned short* WeffL = smem;                  // 64 rows x WSL_LD
    unsigned short* WhhL  = smem + 64 * WSL_LD;    // 64 rows x WSL_LD

    int lane = threadIdx.x & 31, wave = threadIdx.x >> 5;
    int nb = blockIdx.x & 31, mg = blockIdx.x >> 5;
    int n_base = nb * 16;
    int m_base = (mg * 8 + wave) * 16;

    // ---- stage W_eff slice (plain b128 load + ds store) ----
    for (int v = threadIdx.x; v < 64 * 64; v += 256) {
        int r  = v >> 6;                 // 0..63 : gate*16 + rr
        int c8 = (v & 63) << 3;          // 0..504 step 8
        int gate = r >> 4, rr = r & 15;
        const unsigned short* src = Weff + (size_t)(gate * HD + n_base + rr) * HD + c8;
        *(v8bf*)(WeffL + r * WSL_LD + c8) = *(const v8bf*)src;
    }
    // ---- stage W_hh slice via async global->LDS (ASYNCcnt path) ----
    for (int v = threadIdx.x; v < 64 * 64; v += 256) {
        int r  = v >> 6;
        int c8 = (v & 63) << 3;
        int gate = r >> 4, rr = r & 15;
        const unsigned short* src = Whh + (size_t)(gate * HD + n_base + rr) * HD + c8;
        unsigned dlds = (unsigned)(size_t)(WhhL + r * WSL_LD + c8);
        unsigned long long ga = (unsigned long long)(size_t)src;
        asm volatile("global_load_async_to_lds_b128 %0, %1, off"
                     :: "v"(dlds), "v"(ga) : "memory");
    }
    asm volatile("s_wait_asynccnt 0x0" ::: "memory");
    __syncthreads();

    // loop-invariant epilogue constants
    int n = n_base + (lane & 15);
    float bei = b_eff[0 * HD + n], bef = b_eff[1 * HD + n];
    float beg = b_eff[2 * HD + n], beo = b_eff[3 * HD + n];
    float b0i = bias0[0 * HD + n], b0f = bias0[1 * HD + n];
    float b0g = bias0[2 * HD + n], b0o = bias0[3 * HD + n];
    int mrow = m_base + ((lane >> 4) << 3);

    unsigned bcount = 0;
    for (int t = 0; t < T; ++t) {
        const unsigned short* Ah = Hall + (size_t)t * BSZ * HD;        // h_t
        unsigned short*       Ao = Hall + (size_t)(t + 1) * BSZ * HD;  // h_{t+1}
        const unsigned short* Wl = (t == 0) ? WhhL : WeffL;
        float bi  = (t == 0) ? b0i : bei;
        float bf_ = (t == 0) ? b0f : bef;
        float bgv = (t == 0) ? b0g : beg;
        float bo  = (t == 0) ? b0o : beo;

        // gates_t = h_t @ W^T + bias   (K = 512)
        v8f acc0 = {}, acc1 = {}, acc2 = {}, acc3 = {};
#pragma unroll 4
        for (int k0 = 0; k0 < HD; k0 += 32) {
            v16bf a  = load_frag_a(Ah, HD, m_base, k0, lane);
            v16bf b0 = load_frag_b(Wl, WSL_LD,  0, k0, lane);
            v16bf b1 = load_frag_b(Wl, WSL_LD, 16, k0, lane);
            v16bf b2 = load_frag_b(Wl, WSL_LD, 32, k0, lane);
            v16bf b3 = load_frag_b(Wl, WSL_LD, 48, k0, lane);
            acc0 = wmma_bf16(acc0, a, b0);
            acc1 = wmma_bf16(acc1, a, b1);
            acc2 = wmma_bf16(acc2, a, b2);
            acc3 = wmma_bf16(acc3, a, b3);
        }
#pragma unroll
        for (int r = 0; r < 8; ++r) {
            int m = mrow + r;
            float gi = sigm(acc0[r] + bi);
            float gf = sigm(acc1[r] + bf_);
            float gg = tanh_fast(acc2[r] + bgv);
            float go = sigm(acc3[r] + bo);
            float cn = gf * c[(size_t)m * HD + n] + gi * gg;
            float hn = go * tanh_fast(cn);
            c[(size_t)m * HD + n] = cn;
            Ao[(size_t)m * HD + n] = f2bf(hn);
        }
        grid_barrier(bar, ++bcount * GRIDN);   // h_{t+1} visible everywhere
    }
}

// ---- logits: outs = Hall[1..T] @ W_ho^T + b_ho  (51200 x 512 x 512) --------
__global__ void k_logits(const unsigned short* __restrict__ Ha,   // (T*512,512) bf16
                         const unsigned short* __restrict__ Whob, // (512,512) bf16
                         const float* __restrict__ b_ho,
                         float* __restrict__ outs) {
    int lane = threadIdx.x & 31, wave = threadIdx.x >> 5;
    int n_base = blockIdx.x * 16;
    int m_base = (blockIdx.y * 8 + wave) * 16;
    v8f acc = {};
    for (int k0 = 0; k0 < HD; k0 += 32) {
        v16bf a = load_frag_a(Ha, HD, m_base, k0, lane);
        v16bf b = load_frag_b(Whob, HD, n_base, k0, lane);
        acc = wmma_bf16(acc, a, b);
    }
    int n = n_base + (lane & 15);
    float bias = b_ho[n];
    int mrow = m_base + ((lane >> 4) << 3);
#pragma unroll
    for (int r = 0; r < 8; ++r) {
        int m = mrow + r;
        outs[(size_t)m * ED + n] = acc[r] + bias;
    }
}

// ---- log_softmax over rows of 512, in place --------------------------------
__global__ void k_lsm(float* __restrict__ outs) {
    float* row = outs + (size_t)blockIdx.x * ED;
    int t = threadIdx.x;
    float a = row[t], b = row[t + 256];
    __shared__ float red[256];
    red[t] = fmaxf(a, b);
    __syncthreads();
    for (int s = 128; s > 0; s >>= 1) {
        if (t < s) red[t] = fmaxf(red[t], red[t + s]);
        __syncthreads();
    }
    float mx = red[0];
    __syncthreads();
    red[t] = __expf(a - mx) + __expf(b - mx);
    __syncthreads();
    for (int s = 128; s > 0; s >>= 1) {
        if (t < s) red[t] += red[t + s];
        __syncthreads();
    }
    float ls = logf(red[0]);
    row[t]       = a - mx - ls;
    row[t + 256] = b - mx - ls;
}

// ---- host ------------------------------------------------------------------

extern "C" void kernel_launch(void* const* d_in, const int* in_sizes, int n_in,
                              void* d_out, int out_size, void* d_ws, size_t ws_size,
                              hipStream_t stream) {
    const float* img   = (const float*)d_in[0];
    // d_in[1] = lengths (int scalar); T derived from out_size instead.
    const float* W_out = (const float*)d_in[2];
    const float* b_out = (const float*)d_in[3];
    const float* W_ih  = (const float*)d_in[4];
    const float* W_hh  = (const float*)d_in[5];
    const float* b_ih  = (const float*)d_in[6];
    const float* b_hh  = (const float*)d_in[7];
    const float* W_ho  = (const float*)d_in[8];
    const float* b_ho  = (const float*)d_in[9];
    const float* sos   = (const float*)d_in[10];
    float* outs = (float*)d_out;

    int T = out_size / (BSZ * ED);   // = 100

    char* ws = (char*)d_ws;
    size_t off = 0;
    auto carve = [&](size_t bytes) -> void* {
        void* p = ws + off;
        off = (off + bytes + 255) & ~(size_t)255;
        return p;
    };
    unsigned short* Wihb  = (unsigned short*)carve((size_t)GW * ED * 2);     // 2 MB
    unsigned short* Whhb  = (unsigned short*)carve((size_t)GW * HD * 2);     // 2 MB
    unsigned short* Weffb = (unsigned short*)carve((size_t)GW * HD * 2);     // 2 MB
    unsigned short* Woutb = (unsigned short*)carve((size_t)HD * ENCD * 2);   // 2 MB
    unsigned short* Whob  = (unsigned short*)carve((size_t)ED * HD * 2);     // 0.5 MB
    unsigned short* Whot  = (unsigned short*)carve((size_t)ED * HD * 2);     // 0.5 MB
    unsigned short* imgb  = (unsigned short*)carve((size_t)BSZ * ENCD * 2);  // 2 MB
    float*          c     = (float*)carve((size_t)BSZ * HD * 4);             // 1 MB
    float*          b_eff = (float*)carve((size_t)GW * 4);
    float*          bias0 = (float*)carve((size_t)GW * 4);
    unsigned*       bar   = (unsigned*)carve(256);
    unsigned short* Hall  = (unsigned short*)carve((size_t)(T + 1) * BSZ * HD * 2); // ~53 MB
    (void)in_sizes; (void)n_in; (void)ws_size;

    // one-time conversions + folded weights (amortized over the 100 steps)
    k_f2bf<<<(GW * ED + 255) / 256, 256, 0, stream>>>(W_ih, Wihb, GW * ED);
    k_f2bf<<<(GW * HD + 255) / 256, 256, 0, stream>>>(W_hh, Whhb, GW * HD);
    k_f2bf<<<(ED * HD + 255) / 256, 256, 0, stream>>>(W_ho, Whob, ED * HD);
    k_f2bf_t<<<(ED * HD + 255) / 256, 256, 0, stream>>>(W_ho, Whot);
    k_f2bf<<<(HD * ENCD + 255) / 256, 256, 0, stream>>>(W_out, Woutb, HD * ENCD);
    k_f2bf<<<(BSZ * ENCD + 255) / 256, 256, 0, stream>>>(img, imgb, BSZ * ENCD);
    k_bias2<<<(GW + 255) / 256, 256, 0, stream>>>(W_ih, b_ih, b_hh, b_ho, sos,
                                                  b_eff, bias0);
    k_init_bar<<<1, 64, 0, stream>>>(bar);

    dim3 gw(HD / 16, GW / 128);   // (32, 16)
    k_weff<<<gw, 256, 0, stream>>>(Wihb, Whot, W_hh, Weffb);

    dim3 g(HD / 16, BSZ / 128);   // (32, 4)
    k_h0<<<g, 256, 0, stream>>>(imgb, Woutb, b_out, c, Hall);

    // persistent recurrence: 128 blocks, ~130 KB dynamic LDS each
    size_t shmem = (size_t)SMEM_ELEMS * sizeof(unsigned short);
    k_persist<<<GRIDN, 256, shmem, stream>>>(Weffb, Whhb, b_eff, bias0, c,
                                             Hall, T, bar);

    // logits for all steps at full-device WMMA throughput
    dim3 gl(ED / 16, (unsigned)(T * (BSZ / 128)));   // (32, 400)
    k_logits<<<gl, 256, 0, stream>>>(Hall + (size_t)BSZ * HD, Whob, b_ho, outs);

    k_lsm<<<T * BSZ, 256, 0, stream>>>(outs);
}
